// ComposedLoRAModel_16569983828362
// MI455X (gfx1250) — matrix operand order repaired and checked by reading, not compiled, not run on hardware
//
#include <hip/hip_runtime.h>
#include <math.h>

#define E_ 16
#define L_ 4
#define R_ 16
#define DIN 2048
#define DOUT 2048
#define B_ 4
#define K_ 2
#define S_ 2048
#define JR (K_*R_)   // 32 stacked rank columns
#define PITCH 40     // LDS row pitch in bf16 (32 data + 8 pad) -> 80B rows

typedef __bf16 bf16;
typedef __attribute__((ext_vector_type(16))) __bf16 v16bf;
typedef __attribute__((ext_vector_type(8)))  float  v8f;
typedef __attribute__((ext_vector_type(4)))  __bf16 v4bf;

union FragU { v16bf v; uint4 q[2]; };

// A-matrix 16x32 bf16 fragment per ISA layout:
// lane<16 (half=0): q0=K0..7,  q1=K16..23 of row m=lane&15
// lane>=16(half=1): q0=K8..15, q1=K24..31
__device__ __forceinline__ v16bf fragA(const bf16* rowp, int half) {
  FragU f;
  f.q[0] = *reinterpret_cast<const uint4*>(rowp + half * 8);
  f.q[1] = *reinterpret_cast<const uint4*>(rowp + 16 + half * 8);
  return f.v;
}
// B-matrix 32x16 bf16 fragment (same K order as A): lane<16 holds K0..7,K16..23
// of col n=lane&15, lane>=16 holds K8..15,K24..31 (stored [n][k] row-major)
__device__ __forceinline__ v16bf fragB(const bf16* rowp, int half) {
  FragU f;
  f.q[0] = *reinterpret_cast<const uint4*>(rowp + half * 8);
  f.q[1] = *reinterpret_cast<const uint4*>(rowp + 16 + half * 8);
  return f.v;
}
__device__ __forceinline__ v8f wmma_bf16(v16bf a, v16bf b, v8f c) {
  return __builtin_amdgcn_wmma_f32_16x16x32_bf16(false, a, false, b, (short)0, c,
                                                 false, false);
}

// Async global->LDS 16B copy (no VGPR staging, tracked by ASYNCcnt).
// LDS aperture maps addr[31:0] of the generic pointer to the LDS byte offset.
__device__ __forceinline__ void async_copy_b128(void* lds, const void* gptr) {
  unsigned loff = (unsigned)(unsigned long long)lds;
  asm volatile("global_load_async_to_lds_b128 %0, %1, off"
               :
               : "v"(loff), "v"(gptr)
               : "memory");
}
__device__ __forceinline__ void wait_async0() {
  asm volatile("s_wait_asynccnt 0x0" ::: "memory");
}
// Convert float4 -> 4 x bf16 and store 8B to LDS
__device__ __forceinline__ void cvt_store4(bf16* d, float4 f) {
  v4bf h;
  h.x = (bf16)f.x; h.y = (bf16)f.y; h.z = (bf16)f.z; h.w = (bf16)f.w;
  *reinterpret_cast<v4bf*>(d) = h;
}

// ---------------------------------------------------------------- prep: W -> bf16
__global__ __launch_bounds__(256) void prep_w_kernel(const float* __restrict__ W,
                                                     bf16* __restrict__ Wbf) {
  const int n = L_ * DOUT * DIN;
  for (int t = blockIdx.x * blockDim.x + threadIdx.x; t < n;
       t += gridDim.x * blockDim.x)
    Wbf[t] = (bf16)W[t];
}

// --------------------------------------- prep: gather + mask/scale LoRA factors
// Acomb[l][b][j=k*R+r][i] = mask[e,b,r] * A_w[e,l,r,i]          (bf16)
// Bcomb[l][b][o][j]       = score[b,k]  * B_w[e,l,o,r]          (bf16)
__global__ __launch_bounds__(256) void prep_ab_kernel(
    const float* __restrict__ A_w, const float* __restrict__ B_w,
    const float* __restrict__ tks, const float* __restrict__ masks,
    const int* __restrict__ tki, bf16* __restrict__ Acomb,
    bf16* __restrict__ Bcomb) {
  const int NA = L_ * B_ * JR * DIN;  // 1,048,576
  for (int t = blockIdx.x * blockDim.x + threadIdx.x; t < NA;
       t += gridDim.x * blockDim.x) {
    int i = t & (DIN - 1);
    int j = (t >> 11) & (JR - 1);
    int b = (t >> 16) & (B_ - 1);
    int l = t >> 18;
    int k = j >> 4, r = j & (R_ - 1);
    int e = tki[b * K_ + k];
    float m = masks[(e * B_ + b) * R_ + r];
    float a = A_w[(((size_t)e * L_ + l) * R_ + r) * DIN + i];
    Acomb[t] = (bf16)(m * a);
  }
  const int NB = L_ * B_ * DOUT * JR;  // 1,048,576
  for (int t = blockIdx.x * blockDim.x + threadIdx.x; t < NB;
       t += gridDim.x * blockDim.x) {
    int j = t & (JR - 1);
    int o = (t >> 5) & (DOUT - 1);
    int b = (t >> 16) & (B_ - 1);
    int l = t >> 18;
    int k = j >> 4, r = j & (R_ - 1);
    int e = tki[b * K_ + k];
    float v = tks[b * K_ + k] * B_w[(((size_t)e * L_ + l) * DOUT + o) * R_ + r];
    Bcomb[t] = (bf16)v;
  }
}

// ------------------------------------------------------------- scalar losses
__global__ void losses_kernel(const float* __restrict__ masks,
                              const float* __restrict__ scores,
                              float* __restrict__ out2) {
  __shared__ float red[32];
  int t = threadIdx.x;
  float s = 0.f;
  for (int i = t; i < E_ * B_ * R_; i += 32) s += masks[i];
  red[t] = s;
  __syncthreads();
  if (t == 0) {
    float tot = 0.f;
    for (int i = 0; i < 32; ++i) tot += red[i];
    out2[0] = tot / (float)(E_ * B_ * R_);
    float mean[E_];
    for (int e = 0; e < E_; ++e) mean[e] = 0.f;
    for (int b = 0; b < B_; ++b) {
      float mx = -1e30f;
      for (int e = 0; e < E_; ++e) mx = fmaxf(mx, scores[b * E_ + e]);
      float se = 0.f;
      for (int e = 0; e < E_; ++e) se += expf(scores[b * E_ + e] - mx);
      for (int e = 0; e < E_; ++e)
        mean[e] += expf(scores[b * E_ + e] - mx) / se / (float)B_;
    }
    float u = 1.f / (float)E_, bl = 0.f;
    for (int e = 0; e < E_; ++e) bl += u * (logf(u) - logf(mean[e]));
    out2[1] = bl / (float)E_;
  }
}

// --------------------------------------------- stage 1: T[l][b][s][j] = x·Acombᵀ
// rank-32 GEMM per (l,b): M=S, N=32, K=DIN.  4 waves, 128 s-rows per block.
__global__ __launch_bounds__(128) void t_kernel(const float* __restrict__ x,
                                                const bf16* __restrict__ Acomb,
                                                bf16* __restrict__ Tbf) {
  __shared__ bf16 As[2][128 * PITCH];
  __shared__ bf16 Cs[2][JR * PITCH];
  const int tid = threadIdx.x, lane = tid & 31, wave = tid >> 5;
  const int lb = blockIdx.y;
  const int s0 = blockIdx.x * 128;
  bf16* tp = Tbf + (size_t)lb * S_ * JR;
  const int half = lane >> 4, mn = lane & 15;

  // running per-thread staging pointers (advance +32 elements per K-step)
  const int colx = (tid & 7) * 4, rowx = tid >> 3;   // 16 rows/pass, 8 passes
  const float* xa =
      x + (size_t)lb * S_ * DIN + (size_t)(s0 + rowx) * DIN + colx;
  const int chunk = (tid & 3) * 8, crow = tid >> 2;  // 32 rows, 1 pass
  const bf16* ca = Acomb + (size_t)lb * JR * DIN + (size_t)crow * DIN + chunk;
  const int aoff = rowx * PITCH + colx;   // LDS dest offsets (direct GEP only:
  const int coff = crow * PITCH + chunk;  // keep addrspace(3) => ds_store)

  v8f zero = {0.f, 0.f, 0.f, 0.f, 0.f, 0.f, 0.f, 0.f};
  v8f acc[2][2] = {{zero, zero}, {zero, zero}};

  auto load_stage = [&](int buf) {
    float4 f[8];
#pragma unroll
    for (int p = 0; p < 8; ++p)
      f[p] = *reinterpret_cast<const float4*>(xa + (size_t)p * 16 * DIN);
    __builtin_prefetch(xa + 64, 0, 1);  // speculative: OOB dropped by HW
    async_copy_b128(&Cs[buf][coff], ca);
#pragma unroll
    for (int p = 0; p < 8; ++p)
      cvt_store4(&As[buf][aoff + p * 16 * PITCH], f[p]);
    xa += 32;
    ca += 32;
  };

  load_stage(0);
  wait_async0();
  __syncthreads();
  const int NK = DIN / 32;
  for (int ks = 0; ks < NK; ++ks) {
    int cur = ks & 1;
    if (ks + 1 < NK) load_stage(cur ^ 1);
    v16bf a0 = fragA(&As[cur][(wave * 32 + 0 + mn) * PITCH], half);
    v16bf a1 = fragA(&As[cur][(wave * 32 + 16 + mn) * PITCH], half);
    v16bf b0 = fragB(&Cs[cur][(0 + mn) * PITCH], half);
    v16bf b1 = fragB(&Cs[cur][(16 + mn) * PITCH], half);
    acc[0][0] = wmma_bf16(a0, b0, acc[0][0]);
    acc[0][1] = wmma_bf16(a0, b1, acc[0][1]);
    acc[1][0] = wmma_bf16(a1, b0, acc[1][0]);
    acc[1][1] = wmma_bf16(a1, b1, acc[1][1]);
    wait_async0();
    __syncthreads();
  }
#pragma unroll
  for (int mi = 0; mi < 2; ++mi)
#pragma unroll
    for (int ni = 0; ni < 2; ++ni) {
      int colc = ni * 16 + mn;
#pragma unroll
      for (int r = 0; r < 8; ++r) {
        int row = s0 + wave * 32 + mi * 16 + half * 8 + r;
        tp[(size_t)row * JR + colc] = (bf16)acc[mi][ni][r];
      }
    }
}

// ------------------- stage 2: out = x·Wᵀ + T·Bcombᵀ (rank-32 K-epilogue fused)
// 8 waves, block tile 128(s) x 128(o), wave tile 32 x 64, K-step 32.
// W / T / Bcomb tiles are pure bf16 copies -> async global->LDS DMA;
// x tile needs fp32->bf16 conversion -> VGPR staging + ds_store.
__global__ __launch_bounds__(256) void gemm_kernel(
    const float* __restrict__ x, const bf16* __restrict__ Wbf,
    const bf16* __restrict__ Tbf, const bf16* __restrict__ Bcomb,
    float* __restrict__ out) {
  __shared__ bf16 As[2][128 * PITCH];
  __shared__ bf16 Bs[2][128 * PITCH];
  const int tid = threadIdx.x, lane = tid & 31, wave = tid >> 5;
  const int wm = wave & 3, wn = wave >> 2;
  const int lb = blockIdx.z, l = lb >> 2;
  const int s0 = blockIdx.x * 128, o0 = blockIdx.y * 128;
  const bf16* tp = Tbf + (size_t)lb * S_ * JR;
  const bf16* bp = Bcomb + (size_t)lb * DOUT * JR;
  float* op = out + (size_t)lb * S_ * DOUT;
  const int half = lane >> 4, mn = lane & 15;

  // running per-thread staging pointers (advance +32 elements per K-step)
  const int colx = (tid & 7) * 4, rowx = tid >> 3;   // 32 rows/pass, 4 passes
  const float* xa =
      x + (size_t)lb * S_ * DIN + (size_t)(s0 + rowx) * DIN + colx;
  const int chunk = (tid & 3) * 8, browb = tid >> 2; // 64 rows/pass, 2 passes
  const bf16* wa =
      Wbf + (size_t)l * DOUT * DIN + (size_t)(o0 + browb) * DIN + chunk;
  const int aoff = rowx * PITCH + colx;    // LDS dest offsets (direct GEP only)
  const int boff = browb * PITCH + chunk;

  v8f zero = {0.f, 0.f, 0.f, 0.f, 0.f, 0.f, 0.f, 0.f};
  v8f acc[2][4] = {{zero, zero, zero, zero}, {zero, zero, zero, zero}};

  auto load_main = [&](int buf) {
    float4 f[4];
#pragma unroll
    for (int p = 0; p < 4; ++p)
      f[p] = *reinterpret_cast<const float4*>(xa + (size_t)p * 32 * DIN);
    __builtin_prefetch(xa + 64, 0, 1);  // speculative: OOB dropped by HW
    async_copy_b128(&Bs[buf][boff], wa);
    async_copy_b128(&Bs[buf][boff + 64 * PITCH], wa + (size_t)64 * DIN);
#pragma unroll
    for (int p = 0; p < 4; ++p)
      cvt_store4(&As[buf][aoff + p * 32 * PITCH], f[p]);
    xa += 32;
    wa += 32;
  };
  auto load_rank = [&](int buf) {  // rank-32 epilogue tiles (already bf16)
    const int row = tid >> 2;
#pragma unroll
    for (int p = 0; p < 2; ++p) {
      async_copy_b128(&As[buf][(row + p * 64) * PITCH + chunk],
                      tp + (size_t)(s0 + row + p * 64) * JR + chunk);
      async_copy_b128(&Bs[buf][(row + p * 64) * PITCH + chunk],
                      bp + (size_t)(o0 + row + p * 64) * JR + chunk);
    }
  };
  auto compute = [&](int buf) {
    v16bf a[2], b[4];
#pragma unroll
    for (int mi = 0; mi < 2; ++mi)
      a[mi] = fragA(&As[buf][(wm * 32 + mi * 16 + mn) * PITCH], half);
#pragma unroll
    for (int ni = 0; ni < 4; ++ni)
      b[ni] = fragB(&Bs[buf][(wn * 64 + ni * 16 + mn) * PITCH], half);
#pragma unroll
    for (int mi = 0; mi < 2; ++mi)
#pragma unroll
      for (int ni = 0; ni < 4; ++ni)
        acc[mi][ni] = wmma_bf16(a[mi], b[ni], acc[mi][ni]);
  };

  load_main(0);
  wait_async0();
  __syncthreads();
  const int NK = DIN / 32;  // 64 (even -> rank tile lands in buffer 0)
  for (int ks = 0; ks < NK; ++ks) {
    int cur = ks & 1;
    if (ks + 1 < NK)
      load_main(cur ^ 1);
    else
      load_rank(cur ^ 1);
    compute(cur);
    wait_async0();
    __syncthreads();
  }
  compute(0);  // rank-32 LoRA epilogue step

#pragma unroll
  for (int mi = 0; mi < 2; ++mi)
#pragma unroll
    for (int ni = 0; ni < 4; ++ni) {
      int colg = o0 + wn * 64 + ni * 16 + mn;
#pragma unroll
      for (int r = 0; r < 8; ++r) {
        int rowg = s0 + wm * 32 + mi * 16 + half * 8 + r;
        op[(size_t)rowg * DOUT + colg] = acc[mi][ni][r];
      }
    }
}

// ------------------------------------------------------------------ launcher
extern "C" void kernel_launch(void* const* d_in, const int* in_sizes, int n_in,
                              void* d_out, int out_size, void* d_ws,
                              size_t ws_size, hipStream_t stream) {
  const float* x = (const float*)d_in[0];
  const float* W = (const float*)d_in[1];
  const float* A_w = (const float*)d_in[2];
  const float* B_w = (const float*)d_in[3];
  const float* tks = (const float*)d_in[4];
  const float* nm = (const float*)d_in[5];
  const float* als = (const float*)d_in[6];
  const int* tki = (const int*)d_in[7];
  float* out = (float*)d_out;

  // workspace layout (all bf16): W (32MB) | Acomb (2MB) | Bcomb (2MB) | T (2MB)
  char* ws = (char*)d_ws;
  size_t szW = (size_t)L_ * DOUT * DIN * sizeof(bf16);
  size_t szA = (size_t)L_ * B_ * JR * DIN * sizeof(bf16);
  size_t szB = (size_t)L_ * B_ * DOUT * JR * sizeof(bf16);
  bf16* Wbf = (bf16*)ws;
  bf16* Acomb = (bf16*)(ws + szW);
  bf16* Bcomb = (bf16*)(ws + szW + szA);
  bf16* Tbf = (bf16*)(ws + szW + szA + szB);

  prep_w_kernel<<<8192, 256, 0, stream>>>(W, Wbf);
  prep_ab_kernel<<<2048, 256, 0, stream>>>(A_w, B_w, tks, nm, tki, Acomb, Bcomb);
  losses_kernel<<<1, 32, 0, stream>>>(nm, als,
                                      out + (size_t)L_ * B_ * S_ * DOUT);
  t_kernel<<<dim3(S_ / 128, L_ * B_), 128, 0, stream>>>(x, Acomb, Tbf);
  gemm_kernel<<<dim3(S_ / 128, DOUT / 128, L_ * B_), 256, 0, stream>>>(
      x, Wbf, Tbf, Bcomb, out);
}
